// Network_42039139893790
// MI455X (gfx1250) — compile-verified, matrix operand order checked by reading, not run on hardware
//
#include <hip/hip_runtime.h>
#include <hip/hip_bf16.h>

// ---------------------------------------------------------------------------
// MI455X (gfx1250, wave32) implementation of the e3nn-style 3-layer GNN.
//
// Roofline: radial MLPs are ~57 GFLOP of real GEMM -> bf16 WMMA (f32 accum).
// Edge scatter = f32 atomics into ~26MB node arrays (L2-resident, 192MB L2).
// Radial intermediates stored bf16 to keep HBM streams ~1.3GB (~60us @23.3TB/s).
// Sigmoid/SiLU use v_rcp_f32 (fast reciprocal) instead of IEEE division.
// ---------------------------------------------------------------------------

#define NN     50000
#define EE     1000000
#define NGRAPH 64
#define CIN    16
#define CC     32
#define OUTC   100
#define NB     10
#define MAXR   3.5f

typedef __attribute__((ext_vector_type(16))) __bf16 v16bf;
typedef __attribute__((ext_vector_type(8)))  __bf16 bf16x8;
typedef __attribute__((ext_vector_type(8)))  float  v8f;

// fast sigmoid: v_exp_f32 + v_rcp_f32 (avoids the ~14-op IEEE divide sequence)
__device__ __forceinline__ float sigmoidf_(float x) {
    return __builtin_amdgcn_rcpf(1.0f + __expf(-x));
}

// ---------------------------------------------------------------------------
// Stage 0: per-edge geometry -> cutoff-scaled SH (f32 x4) + gaussian basis
// (bf16, padded 10 -> 16 so it loads directly as half of a WMMA A fragment).
// ---------------------------------------------------------------------------
__global__ __launch_bounds__(256) void edge_pre_kernel(
    const float* __restrict__ pos, const int* __restrict__ ei,
    const float* __restrict__ shift, const float* __restrict__ lattice,
    const int* __restrict__ batch, float* __restrict__ e_sh,
    __bf16* __restrict__ e_basis)
{
    int e = blockIdx.x * 256 + threadIdx.x;
    if (e >= EE) return;
    int src = ei[e], dst = ei[EE + e];
    int gb = batch[src];
    float sx = shift[e*3+0], sy = shift[e*3+1], sz = shift[e*3+2];
    const float* L = lattice + gb * 9;
    float vx = pos[dst*3+0] - pos[src*3+0] + sx*L[0] + sy*L[3] + sz*L[6];
    float vy = pos[dst*3+1] - pos[src*3+1] + sx*L[1] + sy*L[4] + sz*L[7];
    float vz = pos[dst*3+2] - pos[src*3+2] + sx*L[2] + sy*L[5] + sz*L[8];
    float r = sqrtf(vx*vx + vy*vy + vz*vz);
    float rinv = __builtin_amdgcn_rcpf(fmaxf(r, 1e-12f));
    float u = 2.0f * (r / MAXR - 1.0f);
    float cut = (u > 0.0f) ? 0.0f
              : ((u < -2.0f) ? 1.0f
                             : 0.5f * (1.0f - __cosf(3.14159265358979323846f * u)));
    const float SQ3 = 1.7320508075688772f;
    float4 shv;
    shv.x = cut;
    shv.y = cut * SQ3 * vx * rinv;
    shv.z = cut * SQ3 * vy * rinv;
    shv.w = cut * SQ3 * vz * rinv;
    *(float4*)(e_sh + (size_t)e*4) = shv;
    const float step  = MAXR / 9.0f;
    const float istep = 9.0f / MAXR;
    const float scale = (1.0f / 1.12f) * 3.1622776601683795f; // sqrt(10)/1.12
    #pragma unroll
    for (int k = 0; k < 10; k++) {
        float d = (r - step * (float)k) * istep;
        e_basis[(size_t)e*16 + k] = (__bf16)(__expf(-d*d) * scale);
    }
    #pragma unroll
    for (int k = 10; k < 16; k++) e_basis[(size_t)e*16 + k] = (__bf16)0.0f;
}

// ---------------------------------------------------------------------------
// Radial MLP (both GEMMs fused per stage):  w = silu(basis @ W0) @ W1
//   GEMM1: (E,16pad32) x (32,112)  -- 7 wmma per 16-edge tile
//   GEMM2: (E,128)    x (128,K2)   -- 4*N2T wmma per tile (K2 = 16*N2T)
// B matrices pre-packed into LDS in WMMA-fragment order (lane-contiguous),
// h tile bounced through per-wave LDS (C-layout -> A-layout conversion).
// A1 global load is software-pipelined one tile ahead to hide VMEM latency.
// ---------------------------------------------------------------------------
template <int N2T>
__global__ __launch_bounds__(128) void radial_mlp_kernel(
    const __bf16* __restrict__ basis, const float* __restrict__ w0,
    const float* __restrict__ w1, __bf16* __restrict__ wout)
{
    constexpr int K2 = N2T * 16;
    __shared__ __align__(16) __bf16 sB1[7 * 512];
    __shared__ __align__(16) __bf16 sB2[N2T * 4 * 512];
    __shared__ __align__(16) __bf16 sH[4 * 16 * 128];   // 4 waves x (16 rows x 128 cols)

    int tid = threadIdx.x;
    // pack W0 (10x100, zero-padded to 32x112) into fragment order
    for (int xel = tid; xel < 7 * 512; xel += 128) {
        int idx = xel & 15, ln = (xel >> 4) & 31, nt = xel >> 9;
        int h2 = ln >> 4;
        int k = (idx < 8) ? (h2*8 + idx) : (16 + h2*8 + (idx - 8));
        int n = nt*16 + (ln & 15);
        float v = (k < NB && n < 100) ? w0[k*100 + n] : 0.0f;
        sB1[xel] = (__bf16)v;
    }
    // pack W1 (100xK2, zero-padded to 128xK2) into fragment order
    for (int xel = tid; xel < N2T * 4 * 512; xel += 128) {
        int idx = xel & 15, ln = (xel >> 4) & 31, f = xel >> 9;
        int ks = f & 3, nt = f >> 2;
        int h2 = ln >> 4;
        int k = ks*32 + ((idx < 8) ? (h2*8 + idx) : (16 + h2*8 + (idx - 8)));
        int n = nt*16 + (ln & 15);
        float v = (k < 100) ? w1[k*K2 + n] : 0.0f;
        sB2[xel] = (__bf16)v;
    }
    int wid = tid >> 5, lane = tid & 31, hi = lane >> 4, lr = lane & 15;
    // zero padded h columns 112..127 (written once; GEMM1 covers 0..111)
    for (int xel = lane; xel < 256; xel += 32) {
        int row = xel >> 4, col = 112 + (xel & 15);
        sH[wid*2048 + row*128 + col] = (__bf16)0.0f;
    }
    __syncthreads();

    const int ETILES = EE / 16;
    const int stride = (int)gridDim.x * 4;
    int tile = blockIdx.x * 4 + wid;
    bf16x8 blo = {};
    if (tile < ETILES)
        blo = *(const bf16x8*)(basis + (size_t)(tile*16 + lr)*16 + hi*8);

    for (; tile < ETILES; tile += stride) {
        // ---- A1 fragment (K 0..15 live, 16..31 zero) ----
        v16bf a1;
        #pragma unroll
        for (int i = 0; i < 8; i++) { a1[i] = blo[i]; a1[8+i] = (__bf16)0.0f; }

        // prefetch next tile's basis fragment (latency hidden by GEMM1+GEMM2)
        int tnext = tile + stride;
        bf16x8 bnext = {};
        if (tnext < ETILES)
            bnext = *(const bf16x8*)(basis + (size_t)(tnext*16 + lr)*16 + hi*8);

        // ---- GEMM1 + SiLU -> sH (bf16) ----
        #pragma unroll
        for (int nt = 0; nt < 7; nt++) {
            const bf16x8* bp = (const bf16x8*)(sB1 + (nt*32 + lane)*16);
            bf16x8 b0 = bp[0], b1 = bp[1];
            v16bf b;
            #pragma unroll
            for (int i = 0; i < 8; i++) { b[i] = b0[i]; b[8+i] = b1[i]; }
            v8f c = {};
            c = __builtin_amdgcn_wmma_f32_16x16x32_bf16(false, a1, false, b,
                                                        (short)0, c, false, false);
            #pragma unroll
            for (int i = 0; i < 8; i++) {
                float xv = c[i];
                sH[wid*2048 + (hi*8 + i)*128 + nt*16 + lr] = (__bf16)(xv * sigmoidf_(xv));
            }
        }
        __builtin_amdgcn_wave_barrier();   // order LDS stores vs loads (same wave)

        // ---- reload h as A fragments (4 K-steps of 32) ----
        v16bf a2[4];
        #pragma unroll
        for (int ks = 0; ks < 4; ks++) {
            const __bf16* hp = sH + wid*2048 + lr*128 + ks*32;
            bf16x8 lo = *(const bf16x8*)(hp + hi*8);
            bf16x8 hh = *(const bf16x8*)(hp + 16 + hi*8);
            #pragma unroll
            for (int i = 0; i < 8; i++) { a2[ks][i] = lo[i]; a2[ks][8+i] = hh[i]; }
        }
        __builtin_amdgcn_wave_barrier();

        // ---- GEMM2 -> wout (bf16) ----
        #pragma unroll
        for (int nt2 = 0; nt2 < N2T; nt2++) {
            v8f acc = {};
            #pragma unroll
            for (int ks = 0; ks < 4; ks++) {
                const bf16x8* bp = (const bf16x8*)(sB2 + ((nt2*4 + ks)*32 + lane)*16);
                bf16x8 b0 = bp[0], b1 = bp[1];
                v16bf b;
                #pragma unroll
                for (int i = 0; i < 8; i++) { b[i] = b0[i]; b[8+i] = b1[i]; }
                acc = __builtin_amdgcn_wmma_f32_16x16x32_bf16(false, a2[ks], false, b,
                                                              (short)0, acc, false, false);
            }
            #pragma unroll
            for (int i = 0; i < 8; i++) {
                wout[(size_t)(tile*16 + hi*8 + i)*K2 + nt2*16 + lr] = (__bf16)acc[i];
            }
        }
        blo = bnext;
    }
}

// ---------------------------------------------------------------------------
// Message kernels: per-edge gather + atomic scatter (1/sqrt(20) folded in).
// Node accumulators (~26MB) are L2-resident -> atomics stay on-chip.
// ---------------------------------------------------------------------------
#define SEGSC 0.22360679774997896f   // 1/sqrt(20)

__global__ __launch_bounds__(256) void message1_kernel(
    const int* __restrict__ ei, const float* __restrict__ e_sh,
    const __bf16* __restrict__ wbuf, const float* __restrict__ x,
    float* __restrict__ acc_s, float* __restrict__ acc_v)
{
    int e = blockIdx.x * 256 + threadIdx.x;
    if (e >= EE) return;
    int src = ei[e], dst = ei[EE + e];
    float4 shv = *(const float4*)(e_sh + (size_t)e*4);
    float sh0 = shv.x, s1x = shv.y, s1y = shv.z, s1z = shv.w;
    const __bf16* w = wbuf + (size_t)e * 32;
    const float* xs = x + (size_t)src * CIN;
    float* as = acc_s + (size_t)dst * 32;
    float* av = acc_v + (size_t)dst * 96;
    #pragma unroll
    for (int c = 0; c < CIN; c++) {
        float xc = xs[c];
        float ws = (float)w[c], wv = (float)w[CIN + c];
        atomicAdd(as + c, SEGSC * ws * xc * sh0);
        float t = SEGSC * wv * xc;
        atomicAdd(av + c*3 + 0, t * s1x);
        atomicAdd(av + c*3 + 1, t * s1y);
        atomicAdd(av + c*3 + 2, t * s1z);
    }
}

__global__ __launch_bounds__(256) void node_update1_kernel(
    const float* __restrict__ x, const float* __restrict__ z,
    const float* __restrict__ acc_s, const float* __restrict__ acc_v,
    const float* __restrict__ sc1, const float* __restrict__ lin1_s,
    const float* __restrict__ lin1_v,
    float* __restrict__ s_out, float* __restrict__ v_out)
{
    __shared__ float w_sc[CIN*64], w_ls[CIN*64], w_lv[CIN*CC];
    for (int i = threadIdx.x; i < CIN*64; i += 256) { w_sc[i] = sc1[i]; w_ls[i] = lin1_s[i]; }
    for (int i = threadIdx.x; i < CIN*CC; i += 256) w_lv[i] = lin1_v[i];
    __syncthreads();
    int n = blockIdx.x * 256 + threadIdx.x;
    if (n >= NN) return;
    float zn = z[n];
    float xz[CIN], as[CIN], av[CIN*3];
    #pragma unroll
    for (int c = 0; c < CIN; c++) { xz[c] = x[(size_t)n*CIN + c] * zn; as[c] = acc_s[(size_t)n*32 + c]; }
    #pragma unroll
    for (int j = 0; j < CIN*3; j++) av[j] = acc_v[(size_t)n*96 + j];
    for (int o = 0; o < CC; o++) {
        float h = 0.f, g = 0.f, v0 = 0.f, v1 = 0.f, v2 = 0.f;
        #pragma unroll
        for (int c = 0; c < CIN; c++) {
            h += xz[c]*w_sc[c*64 + o]      + as[c]*w_ls[c*64 + o];
            g += xz[c]*w_sc[c*64 + CC + o] + as[c]*w_ls[c*64 + CC + o];
            float lw = w_lv[c*CC + o];
            v0 += av[c*3+0]*lw; v1 += av[c*3+1]*lw; v2 += av[c*3+2]*lw;
        }
        float sg = sigmoidf_(g);
        s_out[(size_t)n*CC + o] = h * sigmoidf_(h);
        v_out[(size_t)n*96 + o*3 + 0] = v0 * sg;
        v_out[(size_t)n*96 + o*3 + 1] = v1 * sg;
        v_out[(size_t)n*96 + o*3 + 2] = v2 * sg;
    }
}

__global__ __launch_bounds__(256) void message2_kernel(
    const int* __restrict__ ei, const float* __restrict__ e_sh,
    const __bf16* __restrict__ wbuf, const float* __restrict__ s_in,
    const float* __restrict__ v_in, float* __restrict__ acc_s,
    float* __restrict__ acc_v)
{
    int e = blockIdx.x * 256 + threadIdx.x;
    if (e >= EE) return;
    int src = ei[e], dst = ei[EE + e];
    float4 shv = *(const float4*)(e_sh + (size_t)e*4);
    float sh0 = shv.x, s1x = shv.y, s1y = shv.z, s1z = shv.w;
    const __bf16* w = wbuf + (size_t)e * 160;
    const float* ss = s_in + (size_t)src * CC;
    const float* vs = v_in + (size_t)src * 96;
    float* as = acc_s + (size_t)dst * 32;
    float* av = acc_v + (size_t)dst * 96;
    for (int c = 0; c < CC; c++) {
        float sc_ = ss[c];
        float vx = vs[c*3+0], vy = vs[c*3+1], vz = vs[c*3+2];
        float w1 = (float)w[c],       w2 = (float)w[32 + c], w3 = (float)w[64 + c];
        float w4 = (float)w[96 + c],  w5 = (float)w[128 + c];
        float ms = w1*sc_*sh0 + w2*(vx*s1x + vy*s1y + vz*s1z);
        atomicAdd(as + c, SEGSC * ms);
        float cx = vy*s1z - vz*s1y, cy = vz*s1x - vx*s1z, cz = vx*s1y - vy*s1x;
        atomicAdd(av + c*3 + 0, SEGSC * (w3*sc_*s1x + w4*vx*sh0 + w5*cx));
        atomicAdd(av + c*3 + 1, SEGSC * (w3*sc_*s1y + w4*vy*sh0 + w5*cy));
        atomicAdd(av + c*3 + 2, SEGSC * (w3*sc_*s1z + w4*vz*sh0 + w5*cz));
    }
}

__global__ __launch_bounds__(256) void node_update2_kernel(
    const float* __restrict__ z, const float* __restrict__ acc_s,
    const float* __restrict__ acc_v, const float* __restrict__ sc2_s,
    const float* __restrict__ sc2_v, const float* __restrict__ lin2_s,
    const float* __restrict__ lin2_v,
    float* __restrict__ s_buf,      // in/out (per-node row, in-place safe)
    const float* __restrict__ v_in, float* __restrict__ v_out)
{
    __shared__ float w_ss[CC*64], w_ls[CC*64], w_sv[CC*CC], w_lv[CC*CC];
    for (int i = threadIdx.x; i < CC*64; i += 256) { w_ss[i] = sc2_s[i]; w_ls[i] = lin2_s[i]; }
    for (int i = threadIdx.x; i < CC*CC; i += 256) { w_sv[i] = sc2_v[i]; w_lv[i] = lin2_v[i]; }
    __syncthreads();
    int n = blockIdx.x * 256 + threadIdx.x;
    if (n >= NN) return;
    float zn = z[n];
    float sz[CC], as[CC], gate[CC];
    #pragma unroll
    for (int c = 0; c < CC; c++) {
        sz[c] = s_buf[(size_t)n*CC + c] * zn;
        as[c] = acc_s[(size_t)n*32 + c];
    }
    for (int o = 0; o < CC; o++) {
        float h = 0.f, g = 0.f;
        #pragma unroll
        for (int c = 0; c < CC; c++) {
            h += sz[c]*w_ss[c*64 + o]      + as[c]*w_ls[c*64 + o];
            g += sz[c]*w_ss[c*64 + CC + o] + as[c]*w_ls[c*64 + CC + o];
        }
        gate[o] = sigmoidf_(g);
        s_buf[(size_t)n*CC + o] = h * sigmoidf_(h);
    }
    for (int i = 0; i < 3; i++) {
        float vc[CC], ac[CC];
        #pragma unroll
        for (int c = 0; c < CC; c++) {
            vc[c] = v_in[(size_t)n*96 + c*3 + i];
            ac[c] = acc_v[(size_t)n*96 + c*3 + i];
        }
        for (int o = 0; o < CC; o++) {
            float vh = 0.f;
            #pragma unroll
            for (int c = 0; c < CC; c++)
                vh += vc[c]*w_sv[c*CC + o] + ac[c]*w_lv[c*CC + o];
            v_out[(size_t)n*96 + o*3 + i] = vh * gate[o];
        }
    }
}

__global__ __launch_bounds__(256) void message_o_kernel(
    const int* __restrict__ ei, const float* __restrict__ e_sh,
    const __bf16* __restrict__ wbuf, const float* __restrict__ s_in,
    const float* __restrict__ v_in, float* __restrict__ acc_s)
{
    int e = blockIdx.x * 256 + threadIdx.x;
    if (e >= EE) return;
    int src = ei[e], dst = ei[EE + e];
    float4 shv = *(const float4*)(e_sh + (size_t)e*4);
    float sh0 = shv.x, s1x = shv.y, s1y = shv.z, s1z = shv.w;
    const __bf16* w = wbuf + (size_t)e * 64;
    const float* ss = s_in + (size_t)src * CC;
    const float* vs = v_in + (size_t)src * 96;
    float* as = acc_s + (size_t)dst * 32;
    for (int c = 0; c < CC; c++) {
        float dotv = vs[c*3+0]*s1x + vs[c*3+1]*s1y + vs[c*3+2]*s1z;
        float m = (float)w[c]*ss[c]*sh0 + (float)w[32 + c]*dotv;
        atomicAdd(as + c, SEGSC * m);
    }
}

__global__ __launch_bounds__(256) void node_out_kernel(
    const float* __restrict__ z, const float* __restrict__ s_in,
    const float* __restrict__ acc_s, const int* __restrict__ batch,
    const float* __restrict__ sco, const float* __restrict__ lino_s,
    float* __restrict__ out)
{
    __shared__ float w_sco[CC*OUTC], w_lin[CC*OUTC];
    for (int i = threadIdx.x; i < CC*OUTC; i += 256) { w_sco[i] = sco[i]; w_lin[i] = lino_s[i]; }
    __syncthreads();
    int n = blockIdx.x * 256 + threadIdx.x;
    if (n >= NN) return;
    float zn = z[n];
    float s2[CC], a[CC];
    #pragma unroll
    for (int c = 0; c < CC; c++) {
        s2[c] = s_in[(size_t)n*CC + c] * zn;
        a[c]  = acc_s[(size_t)n*32 + c];
    }
    float* op = out + (size_t)batch[n] * OUTC;
    const float ISN = 0.035777087639996635f;  // 1/sqrt(781.25)
    for (int j = 0; j < OUTC; j++) {
        float o = 0.f;
        #pragma unroll
        for (int c = 0; c < CC; c++)
            o += s2[c]*w_sco[c*OUTC + j] + a[c]*w_lin[c*OUTC + j];
        atomicAdd(op + j, o * ISN);
    }
}

// ---------------------------------------------------------------------------
extern "C" void kernel_launch(void* const* d_in, const int* in_sizes, int n_in,
                              void* d_out, int out_size, void* d_ws, size_t ws_size,
                              hipStream_t stream)
{
    (void)in_sizes; (void)n_in; (void)ws_size;
    const float* pos     = (const float*)d_in[0];
    const float* x       = (const float*)d_in[1];
    const float* z       = (const float*)d_in[2];
    const int*   ei      = (const int*)  d_in[3];
    const float* shift   = (const float*)d_in[4];
    const float* lattice = (const float*)d_in[5];
    const int*   batch   = (const int*)  d_in[6];
    const float* fc1_w0  = (const float*)d_in[7];
    const float* fc1_w1  = (const float*)d_in[8];
    const float* sc1     = (const float*)d_in[9];
    const float* lin1_s  = (const float*)d_in[10];
    const float* lin1_v  = (const float*)d_in[11];
    const float* fc2_w0  = (const float*)d_in[12];
    const float* fc2_w1  = (const float*)d_in[13];
    const float* sc2_s   = (const float*)d_in[14];
    const float* sc2_v   = (const float*)d_in[15];
    const float* lin2_s  = (const float*)d_in[16];
    const float* lin2_v  = (const float*)d_in[17];
    const float* fco_w0  = (const float*)d_in[18];
    const float* fco_w1  = (const float*)d_in[19];
    const float* sco     = (const float*)d_in[20];
    const float* lino_s  = (const float*)d_in[21];

    char* ws = (char*)d_ws;
    __bf16* e_basis = (__bf16*)(ws + 0);            //  32,000,000 B
    float*  e_sh    = (float*) (ws + 32000000ull);  //  16,000,000 B
    __bf16* wbuf    = (__bf16*)(ws + 48000000ull);  // 320,000,000 B (reused/stage)
    float*  acc_s   = (float*) (ws + 368000000ull); //   6,400,000 B
    float*  acc_v   = (float*) (ws + 374400000ull); //  19,200,000 B (contiguous w/ acc_s)
    float*  s_buf   = (float*) (ws + 393600000ull); //   6,400,000 B
    float*  v_a     = (float*) (ws + 400000000ull); //  19,200,000 B
    float*  v_b     = (float*) (ws + 419200000ull); //  19,200,000 B  (total 438.4 MB)

    float* outp = (float*)d_out;
    hipMemsetAsync(outp, 0, (size_t)out_size * sizeof(float), stream);

    const int EB = (EE + 255) / 256;
    const int NBK = (NN + 255) / 256;

    // stage 0: edge features
    edge_pre_kernel<<<EB, 256, 0, stream>>>(pos, ei, shift, lattice, batch, e_sh, e_basis);

    // ---- layer 1 ----
    radial_mlp_kernel<2><<<2048, 128, 0, stream>>>(e_basis, fc1_w0, fc1_w1, wbuf);
    hipMemsetAsync(acc_s, 0, 6400000ull + 19200000ull, stream);
    message1_kernel<<<EB, 256, 0, stream>>>(ei, e_sh, wbuf, x, acc_s, acc_v);
    node_update1_kernel<<<NBK, 256, 0, stream>>>(x, z, acc_s, acc_v, sc1, lin1_s, lin1_v,
                                                 s_buf, v_a);

    // ---- layer 2 ----
    radial_mlp_kernel<10><<<2048, 128, 0, stream>>>(e_basis, fc2_w0, fc2_w1, wbuf);
    hipMemsetAsync(acc_s, 0, 6400000ull + 19200000ull, stream);
    message2_kernel<<<EB, 256, 0, stream>>>(ei, e_sh, wbuf, s_buf, v_a, acc_s, acc_v);
    node_update2_kernel<<<NBK, 256, 0, stream>>>(z, acc_s, acc_v, sc2_s, sc2_v,
                                                 lin2_s, lin2_v, s_buf, v_a, v_b);

    // ---- output layer ----
    radial_mlp_kernel<4><<<2048, 128, 0, stream>>>(e_basis, fco_w0, fco_w1, wbuf);
    hipMemsetAsync(acc_s, 0, 6400000ull, stream);
    message_o_kernel<<<EB, 256, 0, stream>>>(ei, e_sh, wbuf, s_buf, v_b, acc_s);
    node_out_kernel<<<NBK, 256, 0, stream>>>(z, s_buf, acc_s, batch, sco, lino_s, outp);
}
